// MultiHeadedSelfAttention_76201309766376
// MI455X (gfx1250) — 16-bit and 8-bit matrix operand order repaired and checked by reading, compiled, not run on hardware
//
#include <hip/hip_runtime.h>
#include <hip/hip_bf16.h>

typedef _Float16 v16h __attribute__((ext_vector_type(16)));
typedef float    v8f  __attribute__((ext_vector_type(8)));
typedef int      v8i  __attribute__((ext_vector_type(8)));

union HFrag { v16h v; float4 f4[2]; };
union U8i   { v8i  v; int2 d2[4]; int4 d4[2]; };

__device__ __forceinline__ void async_copy_b128(const void* gsrc, void* lds) {
  unsigned lofs = (unsigned)(size_t)lds;
  unsigned long long gaddr = (unsigned long long)(size_t)gsrc;
  asm volatile("global_load_async_to_lds_b128 %0, %1, off"
               :: "v"(lofs), "v"(gaddr) : "memory");
}
__device__ __forceinline__ void wait_async0() {
  asm volatile("s_wait_asynccnt 0x0" ::: "memory");
}

__global__ __launch_bounds__(256) void quant_rows_kernel(
    const float* __restrict__ src, signed char* __restrict__ dst,
    float* __restrict__ scale, int cols, int mode) {
  int row  = blockIdx.x * 8 + (threadIdx.x >> 5);
  int lane = threadIdx.x & 31;
  const float* s = src + (size_t)row * cols;
  float mx = 0.0f;
  for (int i = lane; i < cols; i += 32) mx = fmaxf(mx, fabsf(s[i]));
  #pragma unroll
  for (int d = 16; d >= 1; d >>= 1) mx = fmaxf(mx, __shfl_xor(mx, d, 32));
  float sc = (mode == 0) ? fmaxf(mx * (1.0f / 127.0f), 1e-12f)
                         : fmaxf(mx, 1e-8f) * (1.0f / 127.0f);
  if (lane == 0) scale[row] = sc;
  float inv = 1.0f / sc;
  signed char* d8 = dst + (size_t)row * cols;
  for (int i = lane; i < cols; i += 32) {
    float q = rintf(s[i] * inv);
    q = fminf(fmaxf(q, -128.0f), 127.0f);
    d8[i] = (signed char)(int)q;
  }
}

__global__ __launch_bounds__(256) void qgemm_kernel(
    const signed char* __restrict__ Aq, const signed char* __restrict__ Wq8,
    const float* __restrict__ sA, const float* __restrict__ sW,
    const float* __restrict__ bias, _Float16* __restrict__ out, int vtrans) {
  constexpr int Kd = 2048, Hh = 16, Wd = 128, SS = 2048;
  constexpr int ROWB = 80;
  __shared__ __align__(16) char bpanel[2][64 * ROWB];

  int wv   = threadIdx.x >> 5;
  int lane = threadIdx.x & 31;
  int mblk = blockIdx.x >> 5;
  int nblk = blockIdx.x & 31;
  int m0 = mblk * 128 + wv * 16;
  int n0 = nblk * 64;
  int ml = lane & 15, half = lane >> 4;

  int prow = threadIdx.x >> 2, pchunk = threadIdx.x & 3;
  const signed char* gsrc = Wq8 + (size_t)(n0 + prow) * Kd + pchunk * 16;
  char* lds0 = &bpanel[0][prow * ROWB + pchunk * 16];
  char* lds1 = &bpanel[1][prow * ROWB + pchunk * 16];

  v8i acc0 = {}, acc1 = {}, acc2 = {}, acc3 = {};
  const signed char* arow = Aq + (size_t)(m0 + ml) * Kd + half * 16;

  auto kstep = [&](const char* panel, int k0) {
    __builtin_prefetch(arow + k0 + 512, 0, 1);
    U8i a;
    a.d2[0] = *(const int2*)(arow + k0);
    a.d2[1] = *(const int2*)(arow + k0 + 8);
    a.d2[2] = *(const int2*)(arow + k0 + 32);
    a.d2[3] = *(const int2*)(arow + k0 + 40);
    const char* b0 = panel + ml * ROWB + half * 16;
    U8i bf;
    bf.d4[0] = *(const int4*)(b0);
    bf.d4[1] = *(const int4*)(b0 + 32);
    acc0 = __builtin_amdgcn_wmma_i32_16x16x64_iu8(true, a.v, true, bf.v, acc0, false, false);
    bf.d4[0] = *(const int4*)(b0 + 16 * ROWB);
    bf.d4[1] = *(const int4*)(b0 + 16 * ROWB + 32);
    acc1 = __builtin_amdgcn_wmma_i32_16x16x64_iu8(true, a.v, true, bf.v, acc1, false, false);
    bf.d4[0] = *(const int4*)(b0 + 32 * ROWB);
    bf.d4[1] = *(const int4*)(b0 + 32 * ROWB + 32);
    acc2 = __builtin_amdgcn_wmma_i32_16x16x64_iu8(true, a.v, true, bf.v, acc2, false, false);
    bf.d4[0] = *(const int4*)(b0 + 48 * ROWB);
    bf.d4[1] = *(const int4*)(b0 + 48 * ROWB + 32);
    acc3 = __builtin_amdgcn_wmma_i32_16x16x64_iu8(true, a.v, true, bf.v, acc3, false, false);
  };

  async_copy_b128(gsrc, lds0);
  for (int i = 0; i < Kd / 64; i += 2) {
    wait_async0();
    __syncthreads();
    async_copy_b128(gsrc + (i + 1) * 64, lds1);
    kstep(bpanel[0], i * 64);
    wait_async0();
    __syncthreads();
    if (i + 2 < Kd / 64) async_copy_b128(gsrc + (i + 2) * 64, lds0);
    kstep(bpanel[1], (i + 1) * 64);
  }

  v8i accs[4] = {acc0, acc1, acc2, acc3};
  int sbase = m0 + 8 * half;
  #pragma unroll
  for (int j = 0; j < 4; ++j) {
    int o = n0 + j * 16 + ml;
    float so = sW[o], bo = bias[o];
    int h = o >> 7, w = o & 127;
    #pragma unroll
    for (int r = 0; r < 8; ++r) {
      int srow = sbase + r;
      float y = (float)accs[j][r] * sA[srow] * so + bo;
      int b = srow >> 11, s = srow & 2047;
      size_t idx = vtrans
          ? ((size_t)((b * Hh + h) * Wd + w) * SS + s)
          : ((size_t)((b * Hh + h) * SS + s) * Wd + w);
      out[idx] = (_Float16)y;
    }
  }
}

__global__ __launch_bounds__(256) void attn_kernel(
    const _Float16* __restrict__ Q, const _Float16* __restrict__ Kh,
    const _Float16* __restrict__ Vt, const int* __restrict__ mask,
    float* __restrict__ out) {
  constexpr int SS = 2048, Wd = 128, Dd = 2048;
  constexpr int KROW = 272;
  constexpr int VROW = 80;
  __shared__ __align__(16) char kpan[2][32 * KROW];
  __shared__ __align__(16) char vpan[2][128 * VROW];
  __shared__ __align__(16) _Float16 pbuf[8][16 * 32];

  int t = threadIdx.x;
  int wv = t >> 5, lane = t & 31;
  int ml = lane & 15, half = lane >> 4;
  int bh = blockIdx.x >> 4;
  int mtile = (blockIdx.x & 15) * 8 + wv;
  int b = bh >> 4, h = bh & 15;
  int m0 = mtile * 16;
  const float scale = 0.08838834764831845f;

  int krow = t >> 3, koff = (t & 7) * 32;
  int vrow = t >> 1, voff = (t & 1) * 32;
  const char* kgb = (const char*)(Kh + ((size_t)bh * SS + krow) * Wd) + koff;
  const char* vgb = (const char*)(Vt + ((size_t)bh * Wd + vrow) * SS) + voff;
  auto issue = [&](int t0, int buf) {
    const char* kg = kgb + (size_t)t0 * (Wd * 2);
    const char* vg = vgb + (size_t)t0 * 2;
    char* kl = &kpan[buf][krow * KROW + koff];
    char* vl = &vpan[buf][vrow * VROW + voff];
    async_copy_b128(kg,      kl);
    async_copy_b128(kg + 16, kl + 16);
    async_copy_b128(vg,      vl);
    async_copy_b128(vg + 16, vl + 16);
  };

  HFrag qA[4];
  const _Float16* qrow = Q + ((size_t)bh * SS + m0 + ml) * Wd;
  #pragma unroll
  for (int kk = 0; kk < 4; ++kk) {
    qA[kk].f4[0] = *(const float4*)(qrow + kk * 32 + half * 8);
    qA[kk].f4[1] = *(const float4*)(qrow + kk * 32 + 16 + half * 8);
  }

  v8f accO[8] = {};
  float mrow[8], lrow[8];
  #pragma unroll
  for (int r = 0; r < 8; ++r) { mrow[r] = -3.0e38f; lrow[r] = 0.0f; }
  const int* mk = mask + b * SS;

  auto step = [&](int t0, int buf) {
    v8f sc[2];
    #pragma unroll
    for (int j = 0; j < 2; ++j) {
      v8f c = {};
      const char* kb0 = &kpan[buf][(j * 16 + ml) * KROW + half * 16];
      #pragma unroll
      for (int kk = 0; kk < 4; ++kk) {
        HFrag kb;
        kb.f4[0] = *(const float4*)(kb0 + kk * 64);
        kb.f4[1] = *(const float4*)(kb0 + kk * 64 + 32);
        c = __builtin_amdgcn_wmma_f32_16x16x32_f16(
            false, qA[kk].v, false, kb.v, (short)0, c, false, false);
      }
      float mb = -10000.0f * (1.0f - (float)mk[t0 + j * 16 + ml]);
      #pragma unroll
      for (int r = 0; r < 8; ++r) c[r] = c[r] * scale + mb;
      sc[j] = c;
    }

    float corr[8];
    #pragma unroll
    for (int r = 0; r < 8; ++r) {
      float vmx = fmaxf(sc[0][r], sc[1][r]);
      #pragma unroll
      for (int d = 8; d >= 1; d >>= 1) vmx = fmaxf(vmx, __shfl_xor(vmx, d, 32));
      float mnew = fmaxf(mrow[r], vmx);
      float c0 = __expf(mrow[r] - mnew);
      mrow[r] = mnew;
      float p0 = __expf(sc[0][r] - mnew);
      float p1 = __expf(sc[1][r] - mnew);
      float rs = p0 + p1;
      #pragma unroll
      for (int d = 8; d >= 1; d >>= 1) rs += __shfl_xor(rs, d, 32);
      lrow[r] = lrow[r] * c0 + rs;
      corr[r] = c0;
      sc[0][r] = p0; sc[1][r] = p1;
    }
    #pragma unroll
    for (int wt = 0; wt < 8; ++wt)
      #pragma unroll
      for (int r = 0; r < 8; ++r) accO[wt][r] *= corr[r];

    _Float16* pb = &pbuf[wv][0];
    #pragma unroll
    for (int r = 0; r < 8; ++r) {
      int m = r + 8 * half;
      pb[m * 32 + ml]      = (_Float16)sc[0][r];
      pb[m * 32 + 16 + ml] = (_Float16)sc[1][r];
    }
    asm volatile("s_wait_dscnt 0" ::: "memory");
    HFrag pA;
    const _Float16* pr = pb + ml * 32;
    pA.f4[0] = *(const float4*)(pr + half * 8);
    pA.f4[1] = *(const float4*)(pr + 16 + half * 8);
    asm volatile("" ::: "memory");

    #pragma unroll
    for (int wt = 0; wt < 8; ++wt) {
      const char* vb0 = &vpan[buf][(wt * 16 + ml) * VROW + half * 16];
      HFrag vb;
      vb.f4[0] = *(const float4*)(vb0);
      vb.f4[1] = *(const float4*)(vb0 + 32);
      accO[wt] = __builtin_amdgcn_wmma_f32_16x16x32_f16(
          false, pA.v, false, vb.v, (short)0, accO[wt], false, false);
    }
  };

  issue(0, 0);
  for (int i = 0; i < SS / 32; i += 2) {
    wait_async0();
    __syncthreads();
    issue((i + 1) * 32, 1);
    step(i * 32, 0);
    wait_async0();
    __syncthreads();
    if (i + 2 < SS / 32) issue((i + 2) * 32, 0);
    step((i + 1) * 32, 1);
  }

  #pragma unroll
  for (int r = 0; r < 8; ++r) {
    float inv = 1.0f / lrow[r];
    int s = m0 + r + 8 * half;
    float* orow = out + ((size_t)b * SS + s) * Dd + h * Wd;
    #pragma unroll
    for (int wt = 0; wt < 8; ++wt)
      orow[wt * 16 + ml] = accO[wt][r] * inv;
  }
}

extern "C" void kernel_launch(void* const* d_in, const int* in_sizes, int n_in,
                              void* d_out, int out_size, void* d_ws, size_t ws_size,
                              hipStream_t stream) {
  (void)in_sizes; (void)n_in; (void)out_size; (void)ws_size;
  const float* x    = (const float*)d_in[0];
  const int*   mask = (const int*)d_in[1];
  const float* wq   = (const float*)d_in[2];
  const float* wk   = (const float*)d_in[3];
  const float* wv   = (const float*)d_in[4];
  const float* bq   = (const float*)d_in[5];
  const float* bk   = (const float*)d_in[6];
  const float* bv   = (const float*)d_in[7];
  float* out = (float*)d_out;

  char* p = (char*)d_ws;
  auto take = [&](size_t bytes) {
    char* r = p;
    p += (bytes + 255) & ~(size_t)255;
    return r;
  };
  signed char* xq  = (signed char*)take(4096ull * 2048);
  signed char* w8q = (signed char*)take(2048ull * 2048);
  signed char* w8k = (signed char*)take(2048ull * 2048);
  signed char* w8v = (signed char*)take(2048ull * 2048);
  float* sx  = (float*)take(4096 * sizeof(float));
  float* swq = (float*)take(2048 * sizeof(float));
  float* swk = (float*)take(2048 * sizeof(float));
  float* swv = (float*)take(2048 * sizeof(float));
  const size_t headBytes = 2ull * 16 * 2048 * 128 * sizeof(_Float16);
  _Float16* Qh = (_Float16*)take(headBytes);
  _Float16* Kh = (_Float16*)take(headBytes);
  _Float16* Vt = (_Float16*)take(headBytes);

  quant_rows_kernel<<<512, 256, 0, stream>>>(x,  xq,  sx,  2048, 0);
  quant_rows_kernel<<<256, 256, 0, stream>>>(wq, w8q, swq, 2048, 1);
  quant_rows_kernel<<<256, 256, 0, stream>>>(wk, w8k, swk, 2048, 1);
  quant_rows_kernel<<<256, 256, 0, stream>>>(wv, w8v, swv, 2048, 1);

  qgemm_kernel<<<1024, 256, 0, stream>>>(xq, w8q, sx, swq, bq, Qh, 0);
  qgemm_kernel<<<1024, 256, 0, stream>>>(xq, w8k, sx, swk, bk, Kh, 0);
  qgemm_kernel<<<1024, 256, 0, stream>>>(xq, w8v, sx, swv, bv, Vt, 1);

  attn_kernel<<<512, 256, 0, stream>>>(Qh, Kh, Vt, mask, out);
}
